// CausalSelfAttention_41592463294720
// MI455X (gfx1250) — matrix operand order repaired and checked by reading, not compiled, not run on hardware
//
#include <hip/hip_runtime.h>
#include <hip/hip_bf16.h>

typedef float    v8f  __attribute__((ext_vector_type(8)));
typedef _Float16 v16h __attribute__((ext_vector_type(16)));
typedef _Float16 v8h  __attribute__((ext_vector_type(8)));

#define Bc   8
#define Tc   1024
#define Cc   1024
#define Hc   16
#define HDc  64
#define N3c  3072
#define EVP  (Tc + 64)   // padded embv-transposed row length

// ---------------- WMMA fragment helpers (gfx1250 wave32 layouts) ----------------
__device__ __forceinline__ v16h cat8(v8h a, v8h b) {
  return __builtin_shufflevector(a, b, 0,1,2,3,4,5,6,7,8,9,10,11,12,13,14,15);
}
// A-frag 16x32 f16: caller passes ptr to (row = lane%16) at k-base.
// Lane holds K in [8*hw,8*hw+8) and [16+8*hw,16+8*hw+8).
__device__ __forceinline__ v16h loadA(const _Float16* rowk, int hw) {
  return cat8(*(const v8h*)(rowk + 8*hw), *(const v8h*)(rowk + 16 + 8*hw));
}
// B-frag 32x16 f16: caller passes ptr to column (n = lane%16) data laid out
// contiguously along K. Lane holds K in [8*hw,8*hw+8) and [16+8*hw,16+8*hw+8).
__device__ __forceinline__ v16h loadB(const _Float16* colk, int hw) {
  return cat8(*(const v8h*)(colk + 8*hw), *(const v8h*)(colk + 16 + 8*hw));
}
__device__ __forceinline__ v8f wmma16(v16h a, v16h b, v8f c) {
  return __builtin_amdgcn_wmma_f32_16x16x32_f16(false, a, false, b, (short)0, c, false, false);
}

// ---------------- small conversion kernels ----------------
__global__ void cvt_f32_f16(const float* __restrict__ s, _Float16* __restrict__ d, int n) {
  int i = blockIdx.x * 256 + threadIdx.x;
  if (i < n) d[i] = (_Float16)s[i];
}
// d[n*K + k] = s[k*N + n]  (weight transpose+convert so B-frags are contiguous)
__global__ void tcvt_f32_f16(const float* __restrict__ s, _Float16* __restrict__ d, int K, int N) {
  int i = blockIdx.x * 256 + threadIdx.x;
  if (i < K * N) {
    int k = i / N, n = i % N;
    d[(size_t)n * K + k] = (_Float16)s[i];
  }
}
// evT[dd][r] = embv[r][dd], padded with zeros for r in [T, T+64)
__global__ void evt_build(const float* __restrict__ embv, _Float16* __restrict__ evT) {
  int i = blockIdx.x * 256 + threadIdx.x;
  if (i < HDc * EVP) {
    int dd = i / EVP, r = i % EVP;
    evT[i] = (r < Tc) ? (_Float16)embv[(size_t)r * HDc + dd] : (_Float16)0.0f;
  }
}

// ---------------- generic f16 WMMA GEMM: C = A * Bt^T + bias ----------------
// A: [M,K] f16 row-major.  Bt: [N,K] f16 row-major (pre-transposed B).
// mode 0: Cout(float)[M,N] = acc + bias[n]
// mode 1: qkv scatter -> Q,K nat [B,H,T,HD] f16, V transposed [B,H,HD,T] f16
__global__ __launch_bounds__(256) void gemm_f16(
    const _Float16* __restrict__ A, const _Float16* __restrict__ Bt,
    const float* __restrict__ bias, float* __restrict__ Cout,
    int M, int N, int K, int mode,
    _Float16* __restrict__ Qo, _Float16* __restrict__ Ko, _Float16* __restrict__ Vto)
{
  const int wid  = threadIdx.x >> 5;
  const int lane = threadIdx.x & 31;
  const int lrow = lane & 15;
  const int hw   = lane >> 4;
  const int wm = wid >> 2, wn = wid & 3;          // 2x4 waves -> 128x256 block tile
  const int m0 = blockIdx.x * 128 + wm * 64;
  const int n0 = blockIdx.y * 256 + wn * 64;

  v8f acc[4][4];
#pragma unroll
  for (int i = 0; i < 4; i++)
#pragma unroll
    for (int j = 0; j < 4; j++) acc[i][j] = (v8f){0,0,0,0,0,0,0,0};

  for (int kb = 0; kb < K; kb += 32) {
    v16h af[4], bf[4];
#pragma unroll
    for (int i = 0; i < 4; i++)
      af[i] = loadA(A + (size_t)(m0 + 16*i + lrow) * K + kb, hw);
#pragma unroll
    for (int j = 0; j < 4; j++)
      bf[j] = loadB(Bt + (size_t)(n0 + 16*j + lrow) * K + kb, hw);
#pragma unroll
    for (int i = 0; i < 4; i++)
#pragma unroll
      for (int j = 0; j < 4; j++)
        acc[i][j] = wmma16(af[i], bf[j], acc[i][j]);
  }

#pragma unroll
  for (int i = 0; i < 4; i++) {
#pragma unroll
    for (int j = 0; j < 4; j++) {
      const int nn = n0 + 16*j + lrow;            // D: N = lane%16
      const float bv = bias[nn];
#pragma unroll
      for (int v = 0; v < 8; v++) {
        const int mm = m0 + 16*i + v + 8*hw;      // D: M = v + 8*(lane/16)
        const float val = acc[i][j][v] + bv;
        if (mode == 0) {
          Cout[(size_t)mm * N + nn] = val;
        } else {
          const int which = nn >> 10, rem = nn & 1023;
          const int h = rem >> 6, dd = rem & 63;
          const int b = mm >> 10, t = mm & 1023;
          const size_t bh = (size_t)(b * Hc + h);
          const _Float16 hv = (_Float16)val;
          if (which == 0)      Qo [(bh * Tc + t) * HDc + dd] = hv;
          else if (which == 1) Ko [(bh * Tc + t) * HDc + dd] = hv;
          else                 Vto[(bh * HDc + dd) * Tc + t] = hv;
        }
      }
    }
  }
}

// ---------------- fused causal attention with relative embeddings ----------------
// grid.x = B*H*(T/128); 256 threads = 8 waves; each wave owns 16 query rows.
__global__ __launch_bounds__(256) void attn_rel_kernel(
    const _Float16* __restrict__ Q,  const _Float16* __restrict__ Kt,
    const _Float16* __restrict__ Vt, const _Float16* __restrict__ ekh,
    const _Float16* __restrict__ evT, _Float16* __restrict__ Y)
{
  __shared__ __align__(64) _Float16 plds[8][16][32];   // per-wave P tile (16t x 32s)

  const int wid  = threadIdx.x >> 5;
  const int lane = threadIdx.x & 31;
  const int lrow = lane & 15;
  const int hw   = lane >> 4;
  const int bh   = blockIdx.x >> 3;          // b*H + h
  const int tblk = blockIdx.x & 7;
  const int b = bh >> 4, h = bh & 15;
  const int t0 = tblk * 128 + wid * 16;

  const _Float16* Qbh = Q  + (size_t)bh * Tc * HDc;
  const _Float16* Kbh = Kt + (size_t)bh * Tc * HDc;
  const _Float16* Vbh = Vt + (size_t)bh * HDc * Tc;

  // Q A-fragments (d = 0..31, 32..63), reused all iterations
  const _Float16* qrow = Qbh + (size_t)(t0 + lrow) * HDc;
  const v16h qa0 = loadA(qrow, hw);
  const v16h qa1 = loadA(qrow + 32, hw);

  // Qe[t,r] = q[t,:]·embk[r,:]  computed as WMMA tiles of 16 r values.
  // Tile with rbase rb:
  auto qe_tile = [&](int rbase) -> v8f {
    int rr = rbase + lrow;
    rr = rr < 0 ? 0 : (rr > Tc - 1 ? Tc - 1 : rr);        // clamp (masked rows)
    const _Float16* er = ekh + (size_t)rr * HDc;
    v8f c = (v8f){0,0,0,0,0,0,0,0};
    c = wmma16(qa0, loadB(er, hw), c);
    c = wmma16(qa1, loadB(er + 32, hw), c);
    return c;
  };

  v8f yacc[4];
#pragma unroll
  for (int dt = 0; dt < 4; dt++) yacc[dt] = (v8f){0,0,0,0,0,0,0,0};
  float rowm[8], rowl[8];
#pragma unroll
  for (int v = 0; v < 8; v++) { rowm[v] = -3.0e38f; rowl[v] = 0.0f; }

  const float scale = 0.125f;  // 1/sqrt(64); applied to q*k^T term only

  // qe0 for the first iteration (rbase = t0); carried across iterations since
  // rbase(d0a-32) of iter j == rbase(d0a) of iter j+1.
  v8f qe0 = qe_tile(t0);

  for (int sj = 0; sj <= t0 + 15; sj += 32) {
    const int d0a = t0 - sj;   // >= 0 (sj <= t0 by block alignment)

    const v8f qe1 = qe_tile(d0a - 16);
    const v8f qe2 = qe_tile(d0a - 32);

    // ---- score tiles (two 16-wide halves of the 32-wide s block)
    v8f att[2];
#pragma unroll
    for (int st = 0; st < 2; st++) {
      const int s16 = sj + 16 * st;
      const int d0  = t0 - s16;
      if (s16 > t0 + 15) {                                 // fully masked half-tile
#pragma unroll
        for (int v = 0; v < 8; v++) att[st][v] = -3.0e38f;
      } else {
        const _Float16* krow = Kbh + (size_t)(s16 + lrow) * HDc;
        v8f s = (v8f){0,0,0,0,0,0,0,0};
        s = wmma16(qa0, loadB(krow, hw), s);
        s = wmma16(qa1, loadB(krow + 32, hw), s);
        const v8f qhi = (st == 0) ? qe0 : qe1;             // rbase = d0
        const v8f qlo = (st == 0) ? qe1 : qe2;             // rbase = d0-16
#pragma unroll
        for (int v = 0; v < 8; v++) {
          const int t_loc = v + 8 * hw;
          const int delta = t_loc - lrow;                  // r - d0, in [-15,15]
          const int srcl  = hw * 16 + (delta & 15);
          const float hi = __shfl(qhi[v], srcl, 32);
          const float lo = __shfl(qlo[v], srcl, 32);
          const float a2 = (delta >= 0) ? hi : lo;
          const int r = d0 + delta;
          att[st][v] = (r >= 0) ? (s[v] * scale + a2) : -3.0e38f;
        }
      }
    }
    qe0 = qe2;   // rotate for next s block

    // ---- online softmax update over the 32-wide block
    float alpha[8];
#pragma unroll
    for (int v = 0; v < 8; v++) {
      float mx = fmaxf(att[0][v], att[1][v]);
#pragma unroll
      for (int o = 1; o < 16; o <<= 1) mx = fmaxf(mx, __shfl_xor(mx, o, 32));
      const float mn = fmaxf(rowm[v], mx);
      alpha[v] = __expf(rowm[v] - mn);
      rowm[v] = mn;
    }
    v8f p0, p1;
#pragma unroll
    for (int v = 0; v < 8; v++) {
      p0[v] = __expf(att[0][v] - rowm[v]);
      p1[v] = __expf(att[1][v] - rowm[v]);
      float sum = (float)p0[v] + (float)p1[v];
#pragma unroll
      for (int o = 1; o < 16; o <<= 1) sum += __shfl_xor(sum, o, 32);
      rowl[v] = rowl[v] * alpha[v] + sum;
#pragma unroll
      for (int dt = 0; dt < 4; dt++) yacc[dt][v] *= alpha[v];
    }

    // ---- stage P (f16) in per-wave LDS slice: plds[wid][t_local][s_local]
#pragma unroll
    for (int v = 0; v < 8; v++) {
      const int t_loc = v + 8 * hw;
      plds[wid][t_loc][lrow]      = (_Float16)p0[v];
      plds[wid][t_loc][16 + lrow] = (_Float16)p1[v];
    }

    // ---- y1 += P[16x32] @ V[32x64]   (V stored transposed [HD,T])
    const v16h pa = loadA(&plds[wid][lrow][0], hw);
#pragma unroll
    for (int dt = 0; dt < 4; dt++) {
      const _Float16* vrow = Vbh + (size_t)(dt * 16 + lrow) * Tc + sj;
      yacc[dt] = wmma16(pa, loadB(vrow, hw), yacc[dt]);
    }

    // ---- y2 += Pr[16x64] @ embv[64x64], Pr[t,r] = p[t, t-r]
    int rb = d0a - 31; if (rb < 0) rb = 0;
    rb &= ~15;                                   // keep evT loads 32B-aligned
#pragma unroll
    for (int f = 0; f < 2; f++) {
      const int rbF = rb + 32 * f;
      v16h pr;
#pragma unroll
      for (int i = 0; i < 16; i++) {
        const int r  = rbF + 8 * hw + i + (i >= 8 ? 8 : 0);  // A-frag K mapping
        const int sl = t0 + lrow - r - sj;                   // s_local in block
        _Float16 val = (_Float16)0.0f;
        if (sl >= 0 && sl < 32) val = plds[wid][lrow][sl];
        pr[i] = val;
      }
#pragma unroll
      for (int dt = 0; dt < 4; dt++) {
        const _Float16* erow = evT + (size_t)(dt * 16 + lrow) * EVP + rbF;
        yacc[dt] = wmma16(pr, loadB(erow, hw), yacc[dt]);
      }
    }
  }

  // ---- finalize: y = yacc / rowl, store f16 into [B,T,C] for the output GEMM
  float invl[8];
#pragma unroll
  for (int v = 0; v < 8; v++) invl[v] = 1.0f / rowl[v];
#pragma unroll
  for (int dt = 0; dt < 4; dt++)
#pragma unroll
    for (int v = 0; v < 8; v++) {
      const int t = t0 + v + 8 * hw;
      Y[(size_t)(b * Tc + t) * Cc + h * HDc + dt * 16 + lrow] =
          (_Float16)(yacc[dt][v] * invl[v]);
    }
}

// ---------------- host-side launch ----------------
extern "C" void kernel_launch(void* const* d_in, const int* in_sizes, int n_in,
                              void* d_out, int out_size, void* d_ws, size_t ws_size,
                              hipStream_t stream) {
  const float* x     = (const float*)d_in[0];
  const float* Wattn = (const float*)d_in[1];
  const float* battn = (const float*)d_in[2];
  const float* Wproj = (const float*)d_in[3];
  const float* bproj = (const float*)d_in[4];
  const float* embk  = (const float*)d_in[5];
  const float* embv  = (const float*)d_in[6];
  float* out = (float*)d_out;

  auto au = [](size_t x) { return (x + 255) & ~(size_t)255; };
  char* w = (char*)d_ws;
  _Float16* xh  = (_Float16*)w; w += au(sizeof(_Float16) * (size_t)Bc * Tc * Cc);
  _Float16* WaT = (_Float16*)w; w += au(sizeof(_Float16) * (size_t)N3c * Cc);
  _Float16* WpT = (_Float16*)w; w += au(sizeof(_Float16) * (size_t)Cc * Cc);
  _Float16* ekh = (_Float16*)w; w += au(sizeof(_Float16) * (size_t)Tc * HDc);
  _Float16* evT = (_Float16*)w; w += au(sizeof(_Float16) * (size_t)HDc * EVP);
  _Float16* Qb  = (_Float16*)w; w += au(sizeof(_Float16) * (size_t)Bc * Hc * Tc * HDc);
  _Float16* Kb  = (_Float16*)w; w += au(sizeof(_Float16) * (size_t)Bc * Hc * Tc * HDc);
  _Float16* Vtb = (_Float16*)w; w += au(sizeof(_Float16) * (size_t)Bc * Hc * HDc * Tc);
  _Float16* Yb  = (_Float16*)w; w += au(sizeof(_Float16) * (size_t)Bc * Tc * Cc);

  auto g1 = [](int n) { return dim3((n + 255) / 256); };

  // precision staging
  cvt_f32_f16<<<g1(Bc * Tc * Cc), 256, 0, stream>>>(x, xh, Bc * Tc * Cc);
  tcvt_f32_f16<<<g1(Cc * N3c), 256, 0, stream>>>(Wattn, WaT, Cc, N3c);
  tcvt_f32_f16<<<g1(Cc * Cc), 256, 0, stream>>>(Wproj, WpT, Cc, Cc);
  cvt_f32_f16<<<g1(Tc * HDc), 256, 0, stream>>>(embk, ekh, Tc * HDc);
  evt_build<<<g1(HDc * EVP), 256, 0, stream>>>(embv, evT);

  // qkv = x @ Wattn + battn, scattered to Q/K (nat) and V (transposed)
  gemm_f16<<<dim3((Bc * Tc) / 128, N3c / 256), 256, 0, stream>>>(
      xh, WaT, battn, nullptr, Bc * Tc, N3c, Cc, 1, Qb, Kb, Vtb);

  // fused causal attention with relative position terms
  attn_rel_kernel<<<dim3(Bc * Hc * (Tc / 128)), 256, 0, stream>>>(
      Qb, Kb, Vtb, ekh, evT, Yb);

  // out = y @ Wproj + bproj
  gemm_f16<<<dim3((Bc * Tc) / 128, Cc / 256), 256, 0, stream>>>(
      Yb, WpT, bproj, out, Bc * Tc, Cc, Cc, 0, nullptr, nullptr, nullptr);
}
